// PixelContrastLoss_18751827214498
// MI455X (gfx1250) — matrix operand order repaired and checked by reading, not compiled, not run on hardware
//
#include <hip/hip_runtime.h>
#include <hip/hip_bf16.h>
#include <math.h>

// ---------------- problem constants (from reference) ----------------
#define BB 8
#define CC 19
#define HW (512 * 512)
#define NMAX 4096
#define MAXV 32
#define TEMP_F 0.07f
#define INV_TEMP (1.0f / 0.07f)
#define LOSS_SCALE (0.07f / 0.07f)   // TEMP / BASE_TEMP
#define IGNORE_CLS 0

// ---------------- workspace byte offsets ----------------
#define OFF_HDR      0          // 16 ints: [0]=N [1]=n_view [2]=T [3]=Npad
#define OFF_CNT      256        // cnt_total[152]
#define OFF_EASY     1280       // cnt_easy[152]
#define OFF_SLOTMAP  2304       // slotmap[152] (b*19+c -> slot or -1)
#define OFF_SLOT_B   3328       // per-slot image index
#define OFF_SLOT_Y   4352      // per-slot class id
#define OFF_SLOT_KH  5376
#define OFF_SLOT_KE  6400
#define OFF_SEL      8192       // sel[152*32] pixel indices
#define OFF_ROWLAB   28672      // rowlab[4096] int
#define OFF_LOSS     45056      // float accumulator
#define OFF_CF       65536      // _Float16 cf[4096][32]
#define OFF_PRED     327680     // int predict[B*HW]
#define ZERO_BYTES   65536

typedef __attribute__((ext_vector_type(16))) _Float16 v16h;
typedef __attribute__((ext_vector_type(8)))  float    v8f;

// ---------------- 1) init: zero counters / header / loss ----------------
__global__ void k_init(int* ws) {
    int i = blockIdx.x * blockDim.x + threadIdx.x;
    if (i < ZERO_BYTES / 4) ws[i] = 0;
}

// ---------------- 2) argmax predict + class histograms ----------------
__global__ void k_predict(const float* __restrict__ inp, const int* __restrict__ tgt,
                          int* __restrict__ pred, int* __restrict__ cnt_total,
                          int* __restrict__ cnt_easy) {
    int idx = blockIdx.x * blockDim.x + threadIdx.x;
    if (idx >= BB * HW) return;
    int b = idx / HW;
    int p = idx - b * HW;
    const float* ip = inp + (size_t)b * CC * HW + p;
    float best = ip[0];
    int bi = 0;
#pragma unroll
    for (int c = 1; c < CC; ++c) {
        float v = ip[(size_t)c * HW];
        if (v > best) { best = v; bi = c; }
    }
    pred[idx] = bi;
    int l = tgt[idx];
    if (l >= 0 && l < CC) {
        atomicAdd(&cnt_total[b * CC + l], 1);
        if (bi == l) atomicAdd(&cnt_easy[b * CC + l], 1);
    }
}

// ---------------- 3) serial planner (152 entries, single thread) ----------------
__global__ void k_plan(int* __restrict__ hdr, const int* __restrict__ cnt_total,
                       const int* __restrict__ cnt_easy, int* __restrict__ slotmap,
                       int* __restrict__ sb, int* __restrict__ sy,
                       int* __restrict__ skh, int* __restrict__ ske) {
    if (threadIdx.x != 0 || blockIdx.x != 0) return;
    int T = 0;
    for (int i = 0; i < BB * CC; ++i) {
        int c = i % CC;
        slotmap[i] = -1;
        if (c != IGNORE_CLS && cnt_total[i] > MAXV) T++;
    }
    int nv = 0;
    if (T > 0) { nv = NMAX / T; if (nv > MAXV) nv = MAXV; if (nv < 1) nv = 1; }
    int ptr = 0;
    for (int b = 0; b < BB; ++b) {
        for (int c = 0; c < CC; ++c) {
            int i = b * CC + c;
            if (c == IGNORE_CLS || cnt_total[i] <= MAXV) continue;
            int ne = cnt_easy[i];
            int nh = cnt_total[i] - ne;
            int kh, ke;
            if (2 * nh >= nv && 2 * ne >= nv) { kh = nv / 2; ke = nv - kh; }
            else if (2 * nh >= nv)            { ke = ne;     kh = nv - ke; }
            else if (2 * ne >= nv)            { kh = nh;     ke = nv - kh; }
            else { kh = nh < nv ? nh : nv; ke = nv - kh; if (ke > ne) ke = ne; }
            slotmap[i] = ptr;
            sb[ptr] = b; sy[ptr] = c; skh[ptr] = kh; ske[ptr] = ke;
            ptr++;
        }
    }
    int N = T * nv;
    hdr[0] = N;
    hdr[1] = nv;
    hdr[2] = T;
    hdr[3] = (N + 15) & ~15;
}

// ---------------- 4) deterministic hard/easy selection (block scan) ----------------
__global__ void k_select(const int* __restrict__ tgt, const int* __restrict__ pred,
                         const int* __restrict__ slotmap, const int* __restrict__ sb,
                         const int* __restrict__ sy, const int* __restrict__ skh,
                         const int* __restrict__ ske, int* __restrict__ sel) {
    int entry = blockIdx.x;              // 0..151 => (b, c)
    int a = slotmap[entry];
    if (a < 0) return;
    int b = sb[a], c = sy[a], kh = skh[a], ke = ske[a];
    const int* lab = tgt + (size_t)b * HW;
    const int* prd = pred + (size_t)b * HW;

    __shared__ int sh[256], se[256];
    int tid = threadIdx.x;
    const int CHUNK = HW / 256;          // 1024
    int start = tid * CHUNK;

    int lh = 0, le = 0;
    for (int i = 0; i < CHUNK; ++i) {
        int p = start + i;
        if (lab[p] == c) { if (prd[p] != c) lh++; else le++; }
    }
    sh[tid] = lh; se[tid] = le;
    __syncthreads();
    for (int off = 1; off < 256; off <<= 1) {
        int vh = (tid >= off) ? sh[tid - off] : 0;
        int ve = (tid >= off) ? se[tid - off] : 0;
        __syncthreads();
        sh[tid] += vh; se[tid] += ve;
        __syncthreads();
    }
    int base_h = sh[tid] - lh;
    int base_e = se[tid] - le;

    int rh = 0, re = 0;
    for (int i = 0; i < CHUNK; ++i) {
        int p = start + i;
        if (lab[p] == c) {
            if (prd[p] != c) {
                int rank = base_h + rh;
                if (rank < kh) sel[a * MAXV + rank] = p;
                rh++;
            } else {
                int rank = base_e + re;
                if (rank < ke) sel[a * MAXV + kh + rank] = p;
                re++;
            }
        }
    }
}

// ---------------- 5) softmax gather -> fp16 cf[4096][32], row labels ----------------
__global__ void k_gather(const float* __restrict__ inp, const int* __restrict__ hdr,
                         const int* __restrict__ sb, const int* __restrict__ sy,
                         const int* __restrict__ sel, _Float16* __restrict__ cf,
                         int* __restrict__ rowlab) {
    int r = blockIdx.x * blockDim.x + threadIdx.x;
    if (r >= NMAX) return;
    int N = hdr[0], T = hdr[2];
    _Float16* row = cf + (size_t)r * 32;
    if (r < N && T > 0) {
        int a = r % T;              // row = v*A + a  (matches transpose/reshape)
        int v = r / T;
        int b = sb[a];
        int pix = sel[a * MAXV + v];
        const float* ip = inp + (size_t)b * CC * HW + pix;
        float mx = ip[0];
#pragma unroll
        for (int c = 1; c < CC; ++c) mx = fmaxf(mx, ip[(size_t)c * HW]);
        float s = 0.0f;
        float e[CC];
#pragma unroll
        for (int c = 0; c < CC; ++c) { e[c] = expf(ip[(size_t)c * HW] - mx); s += e[c]; }
        float inv = 1.0f / s;
#pragma unroll
        for (int c = 0; c < CC; ++c) row[c] = (_Float16)(e[c] * inv);
#pragma unroll
        for (int c = CC; c < 32; ++c) row[c] = (_Float16)0.0f;
        rowlab[r] = sy[a];
    } else {
#pragma unroll
        for (int c = 0; c < 32; ++c) row[c] = (_Float16)0.0f;
        rowlab[r] = -1;
    }
}

// ---------------- 6) WMMA Gram + fused contrastive loss ----------------
// 4 waves / workgroup; wave w owns 16-row block (blockIdx.x*4 + w).
// Column chunks of 64 cf rows (4 KB) are async-copied into double-buffered LDS
// (global_load_async_to_lds_b128, ASYNCcnt, s_wait_asynccnt) and shared by all
// 4 waves. WMMA B fragments are read from LDS; A fragments live in VGPRs for
// all three passes (row-max, neg-sum, pos log-prob).
#define COLS_PER_CHUNK 64
#define NCHUNKS (NMAX / COLS_PER_CHUNK)          // 64
#define CHUNK_HALF (COLS_PER_CHUNK * 32)         // 2048 halves = 4 KB

// Each of the 128 threads copies 32 contiguous bytes (2 x b128).
// ISA: INST_OFFSET applies to BOTH the LDS and global address.
__device__ inline void async_copy_chunk(const _Float16* __restrict__ cf, int rowstart,
                                        _Float16* lbuf) {
    unsigned t = threadIdx.x;                       // 0..127
    unsigned byteoff = t * 32u;
    unsigned long long g = (unsigned long long)(cf + (size_t)rowstart * 32);
    unsigned lds = (unsigned)(uintptr_t)lbuf + byteoff;   // low 32 bits = LDS offset
    unsigned voff = byteoff;
    asm volatile("global_load_async_to_lds_b128 %0, %1, %2 offset:0"
                 :: "v"(lds), "v"(voff), "s"(g) : "memory");
    asm volatile("global_load_async_to_lds_b128 %0, %1, %2 offset:16"
                 :: "v"(lds), "v"(voff), "s"(g) : "memory");
}

__device__ inline void wait_async_le2() {
    asm volatile("s_wait_asynccnt 0x2" ::: "memory");
}
__device__ inline void wait_async_le0() {
    asm volatile("s_wait_asynccnt 0x0" ::: "memory");
}

// B fragment from LDS chunk: lane L (half=L/16, n=tt*16+L%16):
//   elems j<8 = chunk[n*32 + half*8 + j], j>=8 = chunk[n*32 + half*8 + 16 + (j-8)]
__device__ inline v8f tile_mm_lds(const _Float16* lbuf, const v16h& afrag,
                                  int tt, int half, int ln) {
    const _Float16* brow = lbuf + (size_t)(tt * 16 + ln) * 32 + half * 8;
    v16h bf;
#pragma unroll
    for (int j = 0; j < 8; ++j) { bf[j] = brow[j]; bf[8 + j] = brow[16 + j]; }
    v8f c = {};
    return __builtin_amdgcn_wmma_f32_16x16x32_f16(false, afrag, false, bf,
                                                  (short)0, c, false, false);
}

__global__ void __launch_bounds__(128)
k_gram_loss(const _Float16* __restrict__ cf, const int* __restrict__ rowlab,
            const int* __restrict__ hdr, float* __restrict__ lossacc) {
    __shared__ __align__(16) _Float16 tileB[2][CHUNK_HALF];   // 2 x 4 KB

    int N = hdr[0];
    int wave = threadIdx.x >> 5;
    int lane = threadIdx.x & 31;
    int half = lane >> 4;
    int ln   = lane & 15;
    int rowbase = (blockIdx.x * 4 + wave) * 16;   // 0..4080, always in-bounds

    // --- A fragment (CDNA5 16-bit A layout), resident across all passes ---
    v16h afrag;
    {
        const _Float16* arow = cf + (size_t)(rowbase + ln) * 32;
        int kb = half * 8;
#pragma unroll
        for (int j = 0; j < 8; ++j) {
            afrag[j]     = arow[kb + j];
            afrag[8 + j] = arow[kb + 16 + j];
        }
    }
    int rlab[8];
#pragma unroll
    for (int r = 0; r < 8; ++r) rlab[r] = rowlab[rowbase + 8 * half + r];

    // ================= pass 1: row max =================
    float rmax[8];
#pragma unroll
    for (int r = 0; r < 8; ++r) rmax[r] = -INFINITY;

    async_copy_chunk(cf, 0, &tileB[0][0]);
    for (int c = 0; c < NCHUNKS; ++c) {
        int p = c & 1;
        if (c + 1 < NCHUNKS) {
            async_copy_chunk(cf, (c + 1) * COLS_PER_CHUNK, &tileB[p ^ 1][0]);
            wait_async_le2();
        } else {
            wait_async_le0();
        }
        __syncthreads();
#pragma unroll
        for (int tt = 0; tt < 4; ++tt) {
            v8f acc = tile_mm_lds(&tileB[p][0], afrag, tt, half, ln);
            int n = c * COLS_PER_CHUNK + tt * 16 + ln;
            if (n < N) {
#pragma unroll
                for (int r = 0; r < 8; ++r)
                    rmax[r] = fmaxf(rmax[r], acc[r] * INV_TEMP);
            }
        }
        __syncthreads();
    }
#pragma unroll
    for (int r = 0; r < 8; ++r)
        for (int m = 8; m >= 1; m >>= 1)
            rmax[r] = fmaxf(rmax[r], __shfl_xor(rmax[r], m, 32));

    // ================= pass 2: negative exp-sum =================
    float negs[8];
#pragma unroll
    for (int r = 0; r < 8; ++r) negs[r] = 0.0f;

    async_copy_chunk(cf, 0, &tileB[0][0]);
    for (int c = 0; c < NCHUNKS; ++c) {
        int p = c & 1;
        if (c + 1 < NCHUNKS) {
            async_copy_chunk(cf, (c + 1) * COLS_PER_CHUNK, &tileB[p ^ 1][0]);
            wait_async_le2();
        } else {
            wait_async_le0();
        }
        __syncthreads();
#pragma unroll
        for (int tt = 0; tt < 4; ++tt) {
            v8f acc = tile_mm_lds(&tileB[p][0], afrag, tt, half, ln);
            int n = c * COLS_PER_CHUNK + tt * 16 + ln;
            if (n < N) {
                int clab = rowlab[n];
#pragma unroll
                for (int r = 0; r < 8; ++r)
                    if (clab != rlab[r])
                        negs[r] += expf(acc[r] * INV_TEMP - rmax[r]);
            }
        }
        __syncthreads();
    }
#pragma unroll
    for (int r = 0; r < 8; ++r)
        for (int m = 8; m >= 1; m >>= 1)
            negs[r] += __shfl_xor(negs[r], m, 32);

    // ================= pass 3: positive log-prob =================
    float psum[8], pcnt[8];
#pragma unroll
    for (int r = 0; r < 8; ++r) { psum[r] = 0.0f; pcnt[r] = 0.0f; }

    async_copy_chunk(cf, 0, &tileB[0][0]);
    for (int c = 0; c < NCHUNKS; ++c) {
        int p = c & 1;
        if (c + 1 < NCHUNKS) {
            async_copy_chunk(cf, (c + 1) * COLS_PER_CHUNK, &tileB[p ^ 1][0]);
            wait_async_le2();
        } else {
            wait_async_le0();
        }
        __syncthreads();
#pragma unroll
        for (int tt = 0; tt < 4; ++tt) {
            v8f acc = tile_mm_lds(&tileB[p][0], afrag, tt, half, ln);
            int n = c * COLS_PER_CHUNK + tt * 16 + ln;
            if (n < N) {
                int clab = rowlab[n];
#pragma unroll
                for (int r = 0; r < 8; ++r) {
                    int m = rowbase + 8 * half + r;
                    if (clab == rlab[r] && n != m && rlab[r] >= 0) {
                        float l = acc[r] * INV_TEMP - rmax[r];
                        psum[r] += l - logf(expf(l) + negs[r]);
                        pcnt[r] += 1.0f;
                    }
                }
            }
        }
        __syncthreads();
    }
#pragma unroll
    for (int r = 0; r < 8; ++r)
        for (int m = 8; m >= 1; m >>= 1) {
            psum[r] += __shfl_xor(psum[r], m, 32);
            pcnt[r] += __shfl_xor(pcnt[r], m, 32);
        }

    if (ln == 0) {
        float acc = 0.0f;
#pragma unroll
        for (int r = 0; r < 8; ++r) {
            int m = rowbase + 8 * half + r;
            if (m < N && rlab[r] >= 0 && pcnt[r] > 0.0f)
                acc += -LOSS_SCALE * (psum[r] / pcnt[r]);
        }
        atomicAdd(lossacc, acc);
    }
}

// ---------------- 7) finalize ----------------
__global__ void k_final(const float* __restrict__ lossacc, const int* __restrict__ hdr,
                        float* __restrict__ out) {
    if (threadIdx.x == 0 && blockIdx.x == 0) {
        int N = hdr[0];
        out[0] = (N > 0) ? (lossacc[0] / (float)N) : 0.0f;
    }
}

// ---------------- host launcher ----------------
extern "C" void kernel_launch(void* const* d_in, const int* in_sizes, int n_in,
                              void* d_out, int out_size, void* d_ws, size_t ws_size,
                              hipStream_t stream) {
    const float* inputs = (const float*)d_in[0];
    const int*   targets = (const int*)d_in[1];
    float* out = (float*)d_out;

    char* ws = (char*)d_ws;
    int*       hdr       = (int*)(ws + OFF_HDR);
    int*       cnt_total = (int*)(ws + OFF_CNT);
    int*       cnt_easy  = (int*)(ws + OFF_EASY);
    int*       slotmap   = (int*)(ws + OFF_SLOTMAP);
    int*       sb        = (int*)(ws + OFF_SLOT_B);
    int*       sy        = (int*)(ws + OFF_SLOT_Y);
    int*       skh       = (int*)(ws + OFF_SLOT_KH);
    int*       ske       = (int*)(ws + OFF_SLOT_KE);
    int*       sel       = (int*)(ws + OFF_SEL);
    int*       rowlab    = (int*)(ws + OFF_ROWLAB);
    float*     lossacc   = (float*)(ws + OFF_LOSS);
    _Float16*  cf        = (_Float16*)(ws + OFF_CF);
    int*       pred      = (int*)(ws + OFF_PRED);

    k_init<<<(ZERO_BYTES / 4 + 255) / 256, 256, 0, stream>>>((int*)ws);
    k_predict<<<(BB * HW + 255) / 256, 256, 0, stream>>>(inputs, targets, pred,
                                                         cnt_total, cnt_easy);
    k_plan<<<1, 1, 0, stream>>>(hdr, cnt_total, cnt_easy, slotmap, sb, sy, skh, ske);
    k_select<<<BB * CC, 256, 0, stream>>>(targets, pred, slotmap, sb, sy, skh, ske, sel);
    k_gather<<<NMAX / 256, 256, 0, stream>>>(inputs, hdr, sb, sy, sel, cf, rowlab);
    k_gram_loss<<<NMAX / (16 * 4), 128, 0, stream>>>(cf, rowlab, hdr, lossacc);
    k_final<<<1, 1, 0, stream>>>(lossacc, hdr, out);
}
